// LstmPredictor_89378269430252
// MI455X (gfx1250) — matrix operand order repaired and checked by reading, not compiled, not run on hardware
//
#include <hip/hip_runtime.h>
#include <hip/hip_bf16.h>

typedef __attribute__((ext_vector_type(16))) __bf16 v16bf;
typedef __attribute__((ext_vector_type(8)))  float  v8f;

#define GATES 1024   // 4*H
#define HDIM  256
#define TENC  20
#define TDEC  15
#define BTILE 16
#define NT    64     // 1024/16 column tiles
#define KT_ENC 10    // (256+64)/32
#define KT_DEC 8     // 256/32

// ---------------------------------------------------------------------------
// Repack fp32 row-major weight matrix (K x 1024), optionally stacked from two
// sources (rows [0,R0) from src0, rows [R0,K) from src1), into bf16 WMMA
// B-fragment order:
//   packed[(nt*KT + kt)*512 + lane*16 + e] = W[kt*32 + (e>=8?16:0) + (lane>=16?8:0) + (e&7)][nt*16 + (lane&15)]
// so each lane of the GEMM loads its 16 bf16 (32B) contiguously.
// ---------------------------------------------------------------------------
__global__ void repack_b(const float* __restrict__ src0,
                         const float* __restrict__ src1,
                         int R0, int K, __bf16* __restrict__ dst) {
  int KT = K >> 5;
  long total = (long)KT * NT * 512;
  for (long id = (long)blockIdx.x * blockDim.x + threadIdx.x; id < total;
       id += (long)gridDim.x * blockDim.x) {
    int  e    = (int)(id & 15);
    int  lane = (int)((id >> 4) & 31);
    long tile = id >> 9;
    int  kt   = (int)(tile % KT);
    int  nt   = (int)(tile / KT);
    int  k    = kt * 32 + ((e & 8) ? 16 : 0) + ((lane >= 16) ? 8 : 0) + (e & 7);
    int  n    = nt * 16 + (lane & 15);
    float v = (k < R0) ? src0[(long)k * GATES + n]
                       : src1[(long)(k - R0) * GATES + n];
    dst[id] = (__bf16)v;
  }
}

// ---------------------------------------------------------------------------
// A-layout store (16-bit A 16x32, ISA 7.12.2): chunk c covers K=32c..32c+31.
// low lanes hold K offsets {0..7, 16..23}, upper lanes {8..15, 24..31}.
// ---------------------------------------------------------------------------
__device__ inline void store_h_bf16(__bf16* aStage, int m, int j, float h) {
  int c    = j >> 5;
  int ko   = j & 31;
  int lane = m + (((ko & 8) != 0) ? 16 : 0);
  int e    = (ko & 7) + ((ko & 16) ? 8 : 0);
  aStage[(c * 32 + lane) * 16 + e] = (__bf16)h;
}

__device__ inline float sigmoid_f(float x) { return 1.f / (1.f + __expf(-x)); }
__device__ inline float tanh_f(float x) {
  x = fminf(fmaxf(x, -15.f), 15.f);
  float e = __expf(2.f * x);
  return (e - 1.f) / (e + 1.f);
}

// ---------------------------------------------------------------------------
// Per-wave GEMM producing the wave's 8 accumulator tiles:
//   tile q = 2*G + ti  ->  gate G, hidden columns [wave*32+ti*16, +16)
//   (n-tile index nt = G*16 + wave*2 + ti)
// A (16 x 32*KT, bf16 WMMA-A layout) read from LDS; B from packed global.
//
// The opaque zero offset (volatile asm) keeps the fragment addresses
// loop-VARIANT w.r.t. the timestep loop (no LICM hoist -> no spill) while
// preserving the GEP chain from the kernel argument, so InferAddressSpaces
// still emits global_load_b128 (saddr + imm offset, LOADcnt only) rather
// than flat_load (which couples LOADcnt+DScnt and needs 64-bit VALU adds).
//
// All 8 B fragments of a kt are loaded as a batch before the 8 independent
// WMMAs, so ~8-16 loads stay in flight to hide L2 latency.
// ---------------------------------------------------------------------------
template <int KT>
__device__ inline void gemm_acc(v8f acc[8], const __bf16* aStageLds,
                                const __bf16* __restrict__ Bpack,
                                int wave, int lane) {
  unsigned long off = 0;
  asm volatile("" : "+s"(off));          // opaque 0: defeats LICM, keeps addrspace
  const __bf16* B = Bpack + off;

#pragma unroll
  for (int q = 0; q < 8; ++q) acc[q] = (v8f){};

#pragma unroll
  for (int kt = 0; kt < KT; ++kt) {
    v16bf b[8];
#pragma unroll
    for (int q = 0; q < 8; ++q) {
      int nt = (q >> 1) * 16 + wave * 2 + (q & 1);
      b[q] = *(const v16bf*)(B + ((long)nt * KT + kt) * 512 + lane * 16);
    }
    v16bf a = *(const v16bf*)(aStageLds + (kt * 32 + lane) * 16);
#pragma unroll
    for (int q = 0; q < 8; ++q)
      acc[q] = __builtin_amdgcn_wmma_f32_16x16x32_bf16(
          false, a, false, b[q], (short)0, acc[q], false, false);
  }
}

// ---------------------------------------------------------------------------
// Fused network kernel: one workgroup = 16 batch rows, whole net in LDS/regs.
// LDS: aStage (10 KB bf16 A-layout: h in chunks 0..7, u_t in chunks 8..9),
//      xcLds (64 KB decoder input context, accumulator-shaped layout),
//      hBuf  (16 KB fp32 h for the output heads).
// Cell state c and all gate pre-activations live purely in VGPRs.
// ---------------------------------------------------------------------------
__global__ void __launch_bounds__(256, 1)
lstm_fused(const float* __restrict__ x,
           const float* __restrict__ w_in_k, const float* __restrict__ w_in_b,
           const float* __restrict__ enc_b,  const float* __restrict__ dec_b,
           const float* __restrict__ mean_k, const float* __restrict__ mean_b,
           const float* __restrict__ lv_k,   const float* __restrict__ lv_b,
           const __bf16* __restrict__ encW,  const __bf16* __restrict__ decKp,
           const __bf16* __restrict__ decRp, float* __restrict__ out) {
  extern __shared__ char smem[];
  __bf16* aStage = (__bf16*)smem;                       // 10*512 bf16 (10 KB)
  float*  xcLds  = (float*)(aStage + KT_ENC * 512);     // 16x1024    (64 KB)
  float*  hBuf   = xcLds + BTILE * GATES;               // 16x256     (16 KB)

  const int  tid     = threadIdx.x;
  const int  wave    = tid >> 5;
  const int  lane    = tid & 31;
  const int  cidx    = lane & 15;
  const int  rbase   = (lane >= 16) ? 8 : 0;
  const long rowBase = (long)blockIdx.x * BTILE;

  // per-lane gate biases: tile q = 2G+ti -> column G*256 + wave*32 + ti*16 + cidx
  float eb[8], db[8];
#pragma unroll
  for (int q = 0; q < 8; ++q) {
    int col = (q >> 1) * 256 + wave * 32 + (q & 1) * 16 + cidx;
    eb[q] = enc_b[col];
    db[q] = dec_b[col];
  }

  float cReg[16];                 // cell state: [ti*8 + r]
#pragma unroll
  for (int i = 0; i < 16; ++i) cReg[i] = 0.f;
  for (int i = tid; i < 8 * 512; i += 256) aStage[i] = (__bf16)0.f;  // h = 0

  v8f acc[8];

  // ================= encoder: z = [h | u_t] @ [enc_rk ; enc_k] + b =========
  for (int t = 0; t < TENC; ++t) {
    // input projection u_t = relu(x_t @ Win + b) -> aStage chunks 8,9
    for (int idx = tid; idx < BTILE * 64; idx += 256) {
      int m = idx >> 6, q = idx & 63;
      const float* xr = x + ((rowBase + m) * TENC + t) * 8;
      float a = w_in_b[q];
#pragma unroll
      for (int f = 0; f < 8; ++f) a += xr[f] * w_in_k[f * 64 + q];
      a = fmaxf(a, 0.f);
      int cc = q >> 5, ko = q & 31;
      int ln = m + (((ko & 8) != 0) ? 16 : 0);
      int e  = (ko & 7) + ((ko & 16) ? 8 : 0);
      aStage[((8 + cc) * 32 + ln) * 16 + e] = (__bf16)a;
    }
    __syncthreads();                       // h(t-1) + u_t writes visible
    gemm_acc<KT_ENC>(acc, aStage, encW, wave, lane);
    __syncthreads();                       // all waves done reading aStage
    // pointwise cell update entirely in registers
#pragma unroll
    for (int ti = 0; ti < 2; ++ti) {
      int j = wave * 32 + ti * 16 + cidx;
#pragma unroll
      for (int r = 0; r < 8; ++r) {
        float zi = acc[0 + ti][r] + eb[0 + ti];
        float zf = acc[2 + ti][r] + eb[2 + ti];
        float zg = acc[4 + ti][r] + eb[4 + ti];
        float zo = acc[6 + ti][r] + eb[6 + ti];
        float c  = sigmoid_f(zf) * cReg[ti * 8 + r] + sigmoid_f(zi) * tanh_f(zg);
        cReg[ti * 8 + r] = c;
        float h = sigmoid_f(zo) * tanh_f(c);
        store_h_bf16(aStage, rbase + r, j, h);
      }
    }
  }
  __syncthreads();

  // ======= decoder input context (RepeatVector): xc = enc_h @ dec_k ========
  gemm_acc<KT_DEC>(acc, aStage, decKp, wave, lane);
  __syncthreads();
#pragma unroll
  for (int q = 0; q < 8; ++q)
#pragma unroll
    for (int r = 0; r < 8; ++r)
      xcLds[((wave * 8 + q) * 8 + r) * 32 + lane] = acc[q][r];
  // reset state for decoder
#pragma unroll
  for (int i = 0; i < 16; ++i) cReg[i] = 0.f;
  for (int i = tid; i < 8 * 512; i += 256) aStage[i] = (__bf16)0.f;

  // ===================== decoder steps + output heads ======================
  for (int p = 0; p < TDEC; ++p) {
    __syncthreads();                       // h / xc / zero writes visible
    gemm_acc<KT_DEC>(acc, aStage, decRp, wave, lane);
    __syncthreads();                       // all waves done reading aStage
#pragma unroll
    for (int ti = 0; ti < 2; ++ti) {
      int j = wave * 32 + ti * 16 + cidx;
#pragma unroll
      for (int r = 0; r < 8; ++r) {
        float zi = acc[0 + ti][r] + db[0 + ti] +
                   xcLds[((wave * 8 + 0 + ti) * 8 + r) * 32 + lane];
        float zf = acc[2 + ti][r] + db[2 + ti] +
                   xcLds[((wave * 8 + 2 + ti) * 8 + r) * 32 + lane];
        float zg = acc[4 + ti][r] + db[4 + ti] +
                   xcLds[((wave * 8 + 4 + ti) * 8 + r) * 32 + lane];
        float zo = acc[6 + ti][r] + db[6 + ti] +
                   xcLds[((wave * 8 + 6 + ti) * 8 + r) * 32 + lane];
        float c  = sigmoid_f(zf) * cReg[ti * 8 + r] + sigmoid_f(zi) * tanh_f(zg);
        cReg[ti * 8 + r] = c;
        float h = sigmoid_f(zo) * tanh_f(c);
        int m = rbase + r;
        store_h_bf16(aStage, m, j, h);
        hBuf[m * HDIM + j] = h;
      }
    }
    __syncthreads();                       // hBuf complete for heads
    if (tid < 64) {   // mean (oi 0,1) and relu(log_var) (oi 2,3)
      int m = tid >> 2, oi = tid & 3;
      int col = oi & 1;
      const float* wk = (oi < 2) ? mean_k : lv_k;
      float a = (oi < 2) ? mean_b[col] : lv_b[col];
#pragma unroll 4
      for (int k = 0; k < HDIM; ++k) a += hBuf[m * HDIM + k] * wk[k * 2 + col];
      if (oi >= 2) a = fmaxf(a, 0.f);
      out[((rowBase + m) * TDEC + p) * 4 + oi] = a;
    }
  }
}

// ---------------------------------------------------------------------------
extern "C" void kernel_launch(void* const* d_in, const int* in_sizes, int n_in,
                              void* d_out, int out_size, void* d_ws, size_t ws_size,
                              hipStream_t stream) {
  const float* x      = (const float*)d_in[0];
  const float* w_in_k = (const float*)d_in[1];
  const float* w_in_b = (const float*)d_in[2];
  const float* enc_k  = (const float*)d_in[3];
  const float* enc_rk = (const float*)d_in[4];
  const float* enc_b  = (const float*)d_in[5];
  const float* dec_k  = (const float*)d_in[6];
  const float* dec_rk = (const float*)d_in[7];
  const float* dec_b  = (const float*)d_in[8];
  const float* mean_k = (const float*)d_in[9];
  const float* mean_b = (const float*)d_in[10];
  const float* lv_k   = (const float*)d_in[11];
  const float* lv_b   = (const float*)d_in[12];
  float* out = (float*)d_out;

  __bf16* encW = (__bf16*)d_ws;                  // 10*64*512 bf16 = 640 KB
  __bf16* decK = encW + (long)KT_ENC * NT * 512; //  8*64*512 bf16 = 512 KB
  __bf16* decR = decK + (long)KT_DEC * NT * 512; //  8*64*512 bf16 = 512 KB

  // repack weights into bf16 WMMA B-fragment order (encoder K stacked 256+64)
  repack_b<<<512, 256, 0, stream>>>(enc_rk, enc_k, 256, 320, encW);
  repack_b<<<512, 256, 0, stream>>>(dec_k,  dec_k, 256, 256, decK);
  repack_b<<<512, 256, 0, stream>>>(dec_rk, dec_rk, 256, 256, decR);

  int B = in_sizes[0] / (TENC * 8);
  int nBlocks = B / BTILE;
  size_t smemBytes = (size_t)(KT_ENC * 512) * sizeof(__bf16)
                   + (size_t)(BTILE * GATES + BTILE * HDIM) * sizeof(float);
  lstm_fused<<<nBlocks, 256, smemBytes, stream>>>(
      x, w_in_k, w_in_b, enc_b, dec_b, mean_k, mean_b, lv_k, lv_b,
      encW, decK, decR, out);
}
